// OMKANfcn_89953795047915
// MI455X (gfx1250) — matrix operand order repaired and checked by reading, not compiled, not run on hardware
//
#include <hip/hip_runtime.h>
#include <hip/hip_bf16.h>
#include <math.h>

typedef __attribute__((ext_vector_type(16))) _Float16 v16h;
typedef __attribute__((ext_vector_type(8)))  _Float16 v8h;
typedef __attribute__((ext_vector_type(8)))  float    v8f;
typedef __attribute__((ext_vector_type(4)))  unsigned u32x4;
typedef __attribute__((ext_vector_type(8)))  unsigned u32x8;

// ---------------------------------------------------------------------------
// WMMA helpers (CDNA5: V_WMMA_F32_16X16X32_F16, wave32)
// ---------------------------------------------------------------------------
__device__ __forceinline__ v8f wmma16(v16h a, v16h b, v8f c) {
  return __builtin_amdgcn_wmma_f32_16x16x32_f16(false, a, false, b, (short)0, c,
                                                false, false);
}

// A fragment: row-major [16][K], stride sA (halfs). ISA layout:
// lanes 0-15 row=lane: K 0..7 (v0..3), 16..23 (v4..7); lanes16-31: K 8..15,24..31
__device__ __forceinline__ v16h load_a(const _Float16* p, int sA, int kbase, int lane) {
  int m  = lane & 15;
  int ko = (lane >> 4) << 3;                 // 0 or 8
  const _Float16* q = p + m * sA + kbase + ko;
  v8h lo = *(const v8h*)q;
  v8h hi = *(const v8h*)(q + 16);
  v16h r;
#pragma unroll
  for (int i = 0; i < 8; i++) { r[i] = lo[i]; r[i + 8] = hi[i]; }
  return r;
}

// B fragment from B^T storage [N][K], stride sB (halfs).
// lanes 0-15 hold K 0..7,16..23, lanes 16-31 hold K 8..15,24..31, col = lane&15.
__device__ __forceinline__ v16h load_bt(const _Float16* p, int sB, int kbase, int lane) {
  int n  = lane & 15;
  int ko = (lane >> 4) << 3;                 // 0 or 8
  const _Float16* q = p + n * sB + kbase + ko;
  v8h lo = *(const v8h*)q;
  v8h hi = *(const v8h*)(q + 16);
  v16h r;
#pragma unroll
  for (int i = 0; i < 8; i++) { r[i] = lo[i]; r[i + 8] = hi[i]; }
  return r;
}

__device__ __forceinline__ float gelu_f(float x) {
  return 0.5f * x * (1.0f + erff(x * 0.70710678118654752f));
}
__device__ __forceinline__ float sigm(float x) { return 1.0f / (1.0f + expf(-x)); }
__device__ __forceinline__ float softplus_f(float t) {
  return (t > 20.0f) ? t : log1pf(expf(t));
}

// ---------------------------------------------------------------------------
// Tensor Data Mover: issue a load described by D# group0 (built here) and
// group1 (prebuilt).  2-operand form => VADDR2/VADDR3 = NULL (<=2D tensors).
// Caller must be a single wave (TDM ignores EXEC; use a scalar branch).
// ---------------------------------------------------------------------------
__device__ __forceinline__ void tdm_load(unsigned lds_addr,
                                         unsigned long long gaddr, u32x8 g1) {
  u32x4 g0;
  g0[0] = 1u;                                             // count=1, user D#
  g0[1] = (unsigned)__builtin_amdgcn_readfirstlane((int)lds_addr);
  g0[2] = (unsigned)__builtin_amdgcn_readfirstlane((int)(unsigned)gaddr);
  g0[3] = ((unsigned)__builtin_amdgcn_readfirstlane((int)(unsigned)(gaddr >> 32))
           & 0x1FFFFFFu) | 0x80000000u;                   // type=2 ("image")
  asm volatile("tensor_load_to_lds %0, %1" :: "s"(g0), "s"(g1) : "memory");
}

// ---------------------------------------------------------------------------
// f32 -> f16 conversion
// ---------------------------------------------------------------------------
__global__ void k_cvt_h(const float* __restrict__ s, _Float16* __restrict__ d, int n) {
  int i = blockIdx.x * blockDim.x + threadIdx.x;
  if (i < n) d[i] = (_Float16)s[i];
}

// ---------------------------------------------------------------------------
// Fused encoder: conv1(1x8,SAME)+BN+GELU tile -> WMMA K=8192 GEMM with we2
//                -> BN+GELU -> x_src (f16) [B*T][64]
// B tiles (64x32 f16 = 4KB) are TDM-prefetched into double-buffered LDS,
// one K-step ahead, synchronized with TENSORcnt + workgroup barrier.
// grid (32, 16), 128 threads (4 waves; wave w owns e-tile w)
// ---------------------------------------------------------------------------
__global__ __launch_bounds__(128) void k_encoder(
    const float* __restrict__ x,      // [32][256][32]
    const float* __restrict__ we1,    // [256][8]
    const float* __restrict__ be1,
    const float* __restrict__ g1, const float* __restrict__ b1,
    const float* __restrict__ m1, const float* __restrict__ v1,
    const _Float16* __restrict__ we2t,  // [64][8192]  (e-major, k=c*32+f)
    const float* __restrict__ be2,
    const float* __restrict__ g2, const float* __restrict__ b2,
    const float* __restrict__ m2, const float* __restrict__ v2,
    _Float16* __restrict__ xsh)       // [8192][64]
{
  __shared__ __align__(16) _Float16 actT[16 * 32];       // [t_local][f]
  __shared__ __align__(16) _Float16 bbuf[2][64 * 32];    // double-buffered B tile
  int b   = blockIdx.x;
  int t0  = blockIdx.y * 16;
  int tid = threadIdx.x;
  int lane = tid & 31, wave = tid >> 5;
  bool issuer = (__builtin_amdgcn_readfirstlane((int)threadIdx.x) < 32);

  // D# group1 for a 2D tile: data_size=2B, tensor 8192x64, tile 32x64,
  // dim0 stride 8192 (halfs)
  u32x8 g1t;
  g1t[0] = 1u << 16;          // data_size=1 (2 bytes), mask=0
  g1t[1] = 0x2000u << 16;     // tensor_dim0[15:0] = 8192
  g1t[2] = 64u << 16;         // tensor_dim1 = 64
  g1t[3] = 32u << 16;         // tile_dim0 = 32
  g1t[4] = 64u;               // tile_dim1 = 64
  g1t[5] = 8192u;             // tensor_dim0_stride = 8192
  g1t[6] = 0u;
  g1t[7] = 0u;

  unsigned lds_b = (unsigned)(unsigned long long)(&bbuf[0][0]);
  unsigned long long gb = (unsigned long long)we2t;

  if (issuer) tdm_load(lds_b, gb, g1t);                  // prefetch c=0

  v8f acc = {};
  for (int c = 0; c < 256; c++) {
    float inv = rsqrtf(v1[c] + 1e-5f);
    float gg = g1[c], bb = b1[c], mm = m1[c], bias = be1[c];
    // 16x32 activation tile for this c (overlaps in-flight TDM)
    for (int e = tid; e < 512; e += 128) {
      int tl = e >> 5, f = e & 31;
      int t = t0 + tl;
      float s = bias;
#pragma unroll
      for (int j = 0; j < 8; j++) {
        int ts = t + j - 3;
        if (ts >= 0 && ts < 256) s += x[b * 8192 + ts * 32 + f] * we1[c * 8 + j];
      }
      float a = (s - mm) * inv * gg + bb;
      actT[tl * 32 + f] = (_Float16)gelu_f(a);
    }
    if (issuer) __builtin_amdgcn_s_wait_tensorcnt(0);    // tile c landed
    __syncthreads();
    if (issuer && (c + 1) < 256)                          // prefetch c+1
      tdm_load(lds_b + (unsigned)(((c + 1) & 1) * 64 * 32 * 2),
               gb + (unsigned long long)(c + 1) * 64ull, g1t);
    v16h A  = load_a(actT, 32, 0, lane);
    v16h Bf = load_bt(&bbuf[c & 1][0] + wave * 16 * 32, 32, 0, lane);
    acc = wmma16(A, Bf, acc);
    __syncthreads();
  }
  // epilogue: bias + BN + GELU, store f16
  int n0 = wave * 16;
#pragma unroll
  for (int i = 0; i < 8; i++) {
    int m = i + ((lane >> 4) << 3);
    int e = n0 + (lane & 15);
    float y = acc[i] + be2[e];
    y = (y - m2[e]) * rsqrtf(v2[e] + 1e-5f) * g2[e] + b2[e];
    xsh[(size_t)(b * 256 + t0 + m) * 64 + e] = (_Float16)gelu_f(y);
  }
}

// ---------------------------------------------------------------------------
// gates_pre = x_src @ w_ih^T + (b_ih + b_hh)   [8192][512]
// grid (512, 8), 128 threads (4 waves; wave owns one n-tile)
// ---------------------------------------------------------------------------
__global__ __launch_bounds__(128) void k_gates_in(
    const _Float16* __restrict__ xsh,  // [8192][64]
    const _Float16* __restrict__ wih,  // [512][64] (already B^T)
    const float* __restrict__ b_ih, const float* __restrict__ b_hh,
    float* __restrict__ gates)         // [8192][512]
{
  int lane = threadIdx.x & 31, wave = threadIdx.x >> 5;
  int mt = blockIdx.x;
  int nt = blockIdx.y * 4 + wave;
  const _Float16* Ap = xsh + (size_t)mt * 16 * 64;
  const _Float16* Bp = wih + (size_t)nt * 16 * 64;
  v8f acc = {};
  acc = wmma16(load_a(Ap, 64, 0, lane),  load_bt(Bp, 64, 0, lane),  acc);
  acc = wmma16(load_a(Ap, 64, 32, lane), load_bt(Bp, 64, 32, lane), acc);
#pragma unroll
  for (int i = 0; i < 8; i++) {
    int m = mt * 16 + i + ((lane >> 4) << 3);
    int n = nt * 16 + (lane & 15);
    gates[(size_t)m * 512 + n] = acc[i] + b_ih[n] + b_hh[n];
  }
}

// ---------------------------------------------------------------------------
// Persistent LSTM: single workgroup, 8 waves, 256 steps.
// w_hh (512x128 f16 = 128KB) is TDM-parked in LDS once at kernel start.
// Per step: G = gates_pre[:,t,:] + h @ w_hh^T (WMMA), elementwise update.
// ---------------------------------------------------------------------------
__global__ __launch_bounds__(256) void k_lstm(
    const float* __restrict__ gates,   // [b*256+t][512]
    const _Float16* __restrict__ whh,  // [512][128] (B^T)
    float* __restrict__ hseq)          // [32][256][128]
{
  __shared__ __align__(16) _Float16 whh_l[512 * 128];  // 128 KB weight cache
  __shared__ __align__(16) _Float16 hbuf[32 * 128];    // h state (f16, A layout)
  __shared__ float Gs[32 * 512];                       // gate pre-activations
  int tid = threadIdx.x;
  int lane = tid & 31, wave = tid >> 5;
  bool issuer = (__builtin_amdgcn_readfirstlane((int)threadIdx.x) < 32);

  if (issuer) {
    // 1-D copy: data_size=8B, 16384 units, single row
    u32x8 g1t;
    g1t[0] = 3u << 16;          // data_size=3 (8 bytes)
    g1t[1] = 0x4000u << 16;     // tensor_dim0[15:0] = 16384
    g1t[2] = 0u;
    g1t[3] = 0x4000u << 16;     // tile_dim0 = 16384
    g1t[4] = 0u;                // tile_dim1 unused
    g1t[5] = 16384u;            // dim0 stride
    g1t[6] = 0u; g1t[7] = 0u;
    tdm_load((unsigned)(unsigned long long)(&whh_l[0]),
             (unsigned long long)whh, g1t);
  }

  float cst[16];
#pragma unroll
  for (int r = 0; r < 16; r++) cst[r] = 0.0f;
  for (int i = tid; i < 32 * 128; i += 256) hbuf[i] = (_Float16)0.0f;
  if (issuer) __builtin_amdgcn_s_wait_tensorcnt(0);
  __syncthreads();

  for (int t = 0; t < 256; t++) {
    // ---- G = h @ w_hh^T : 64 tiles (2 m-tiles x 32 n-tiles), 8 per wave ----
    for (int q = 0; q < 8; q++) {
      int tile = wave * 8 + q;
      int mt = tile >> 5, nt = tile & 31;
      const _Float16* Ap = hbuf + mt * 16 * 128;
      const _Float16* Bp = whh_l + nt * 16 * 128;
      v8f acc = {};
#pragma unroll
      for (int ks = 0; ks < 4; ks++)
        acc = wmma16(load_a(Ap, 128, ks * 32, lane),
                     load_bt(Bp, 128, ks * 32, lane), acc);
#pragma unroll
      for (int i = 0; i < 8; i++) {
        int m = mt * 16 + i + ((lane >> 4) << 3);
        int n = nt * 16 + (lane & 15);
        Gs[m * 512 + n] = acc[i] + gates[((size_t)m * 256 + t) * 512 + n];
      }
    }
    __syncthreads();
    // ---- elementwise update: thread owns 16 (b,h) pairs ----
#pragma unroll
    for (int r = 0; r < 16; r++) {
      int flat = r * 256 + tid;
      int b = flat >> 7, h = flat & 127;
      float gi = Gs[b * 512 + h];
      float gf = Gs[b * 512 + 128 + h];
      float gg = Gs[b * 512 + 256 + h];
      float go = Gs[b * 512 + 384 + h];
      float c = sigm(gf) * cst[r] + sigm(gi) * tanhf(gg);
      cst[r] = c;
      float hv = sigm(go) * tanhf(c);
      hbuf[b * 128 + h] = (_Float16)hv;
      hseq[((size_t)b * 256 + t) * 128 + h] = hv;
    }
    __syncthreads();
  }
}

// ---------------------------------------------------------------------------
// sim = exp(-5.5*(1 - hn @ hn^T)), diag zeroed.  One block per batch.
// ---------------------------------------------------------------------------
__global__ __launch_bounds__(256) void k_sim(
    const float* __restrict__ hseq,  // [32][256][128]
    float* __restrict__ sim)         // [32][256][256]
{
  __shared__ __align__(16) _Float16 hn[256 * 128];
  int b = blockIdx.x;
  int tid = threadIdx.x;
  int lane = tid & 31, wave = tid >> 5;
  const float* hb = hseq + (size_t)b * 256 * 128;

  {
    int t = tid;  // 256 threads, one row each
    float s = 0.0f;
    for (int k = 0; k < 128; k++) { float v = hb[t * 128 + k]; s += v * v; }
    float d = 1.0f / fmaxf(sqrtf(s), 1e-8f);
    for (int k = 0; k < 128; k++) hn[t * 128 + k] = (_Float16)(hb[t * 128 + k] * d);
  }
  __syncthreads();

  float* sb = sim + (size_t)b * 65536;
  for (int q = 0; q < 32; q++) {
    int tile = wave * 32 + q;
    int mt = tile >> 4, nt = tile & 15;
    const _Float16* Ap = hn + mt * 16 * 128;
    const _Float16* Bp = hn + nt * 16 * 128;  // B^T == hn (symmetric product)
    v8f acc = {};
#pragma unroll
    for (int ks = 0; ks < 4; ks++)
      acc = wmma16(load_a(Ap, 128, ks * 32, lane),
                   load_bt(Bp, 128, ks * 32, lane), acc);
#pragma unroll
    for (int i = 0; i < 8; i++) {
      int mg = mt * 16 + i + ((lane >> 4) << 3);
      int ng = nt * 16 + (lane & 15);
      float v = (mg == ng) ? 0.0f : expf(-5.5f * (1.0f - acc[i]));
      sb[mg * 256 + ng] = v;
    }
  }
}

// ---------------------------------------------------------------------------
// 2-D prefix sums, segmentation score, cut, corr/last.  One block per batch.
// ---------------------------------------------------------------------------
__global__ __launch_bounds__(256) void k_cluster(
    float* __restrict__ sim,          // [32][256][256]  (destroyed)
    const float* __restrict__ hseq,   // [32][256][128]
    float* __restrict__ last,         // [32][128]
    float* __restrict__ corr)         // [32][128]
{
  __shared__ float Rl[256], Cl[256], Sii[256];
  __shared__ float bval[256];
  __shared__ int   bidx[256];
  __shared__ int   cut_s;
  int b = blockIdx.x, tid = threadIdx.x;
  float* S = sim + (size_t)b * 65536;

  // pass 1: per-row prefix along s (thread = row)
  {
    float run = 0.0f;
    for (int s = 0; s < 256; s++) { run += S[tid * 256 + s]; S[tid * 256 + s] = run; }
  }
  __syncthreads();
  // pass 2: running column sums (thread = column)
  {
    float run = 0.0f;
    for (int t = 0; t < 256; t++) {
      run += S[t * 256 + tid];
      if (t == tid) Sii[tid] = run;
      if (tid == 255) Cl[t] = run;
    }
    Rl[tid] = run;
  }
  __syncthreads();
  float tot = Rl[255];
  // dist over k = 0..254 (cut index = k+1)
  {
    float v = -1e30f;
    if (tid < 255) {
      float i_f = (float)(tid + 1);
      float Tm  = 255.0f - (float)tid;
      float sii = Sii[tid], r = Rl[tid], c = Cl[tid];
      v = sii / (i_f * i_f) + (tot - r - c + sii) / (Tm * Tm)
          - (c - sii) / (i_f * Tm) - (r - sii) / (i_f * Tm);
    }
    bval[tid] = v; bidx[tid] = tid;
  }
  __syncthreads();
  for (int o = 128; o > 0; o >>= 1) {
    if (tid < o) {
      float a = bval[tid], c = bval[tid + o];
      int ai = bidx[tid], ci = bidx[tid + o];
      if (c > a || (c == a && ci < ai)) { bval[tid] = c; bidx[tid] = ci; }
    }
    __syncthreads();
  }
  if (tid == 0) cut_s = bidx[0] + 1;
  __syncthreads();
  int cut = cut_s;
  if (tid < 128) {
    const float* hb = hseq + (size_t)b * 256 * 128;
    float s = 0.0f;
    for (int t = 0; t < cut; t++) s += hb[t * 128 + tid];
    corr[b * 128 + tid] = s / (float)cut;
    last[b * 128 + tid] = hb[255 * 128 + tid];
  }
}

// ---------------------------------------------------------------------------
// KAN: out[b][j] = sum_i ( W2[i,j] . silu(W1[i,j] @ feat(x[b,i]) + B1) + B2 )
// grid (32, dout), 128 threads
// ---------------------------------------------------------------------------
__global__ __launch_bounds__(128) void k_kan(
    const float* __restrict__ xin, int din, int dout,
    const float* __restrict__ W1, const float* __restrict__ W2,
    const float* __restrict__ B1, const float* __restrict__ B2,
    float* __restrict__ out)
{
  int b = blockIdx.x, j = blockIdx.y, tid = threadIdx.x;
  float s = 0.0f;
  for (int i = tid; i < din; i += 128) {
    float xv = xin[b * din + i];
    float feat[7];
    feat[0] = xv;
    feat[1] = sinf(xv);        feat[2] = sinf(2.0f * xv); feat[3] = sinf(4.0f * xv);
    feat[4] = cosf(xv);        feat[5] = cosf(2.0f * xv); feat[6] = cosf(4.0f * xv);
    size_t ij = (size_t)i * dout + j;
    const float* w1 = W1 + ij * 224;
    const float* bb = B1 + ij * 32;
    const float* w2 = W2 + ij * 32;
    float accj = B2[ij];
    for (int h = 0; h < 32; h++) {
      float a = bb[h];
#pragma unroll
      for (int f = 0; f < 7; f++) a += feat[f] * w1[h * 7 + f];
      accj += w2[h] * (a * sigm(a));  // silu
    }
    s += accj;
  }
  __shared__ float red[128];
  red[tid] = s; __syncthreads();
  for (int o = 64; o > 0; o >>= 1) { if (tid < o) red[tid] += red[tid + o]; __syncthreads(); }
  if (tid == 0) out[b * dout + j] = red[0];
}

// ---------------------------------------------------------------------------
// glue: z / concat kernels
// ---------------------------------------------------------------------------
__global__ void k_zcat(const float* mu, const float* st, const float* eps1,
                       const float* last, float* x_z) {
  int i = blockIdx.x * blockDim.x + threadIdx.x;
  if (i < 4096) {
    int b = i >> 7, h = i & 127;
    float z = mu[i] + softplus_f(st[i] - 5.0f) * eps1[i];
    x_z[b * 256 + h] = last[i];
    x_z[b * 256 + 128 + h] = z;
  }
}
__global__ void k_x1cat(const float* muc, const float* stdc, const float* eps2,
                        const float* x2v, float* x_all) {
  int i = blockIdx.x * blockDim.x + threadIdx.x;
  if (i < 4096) {
    int b = i >> 7, h = i & 127;
    x_all[b * 256 + h] = muc[i] + softplus_f(stdc[i] - 5.0f) * eps2[i];
    x_all[b * 256 + 128 + h] = x2v[i];
  }
}

// ---------------------------------------------------------------------------
// causal conv1d (+bias+BN+GELU).  in [B][Cin][32] -> out [B][Cout][32]
// grid (32, Cout/8), 256 threads (8 out-ch x 32 positions)
// ---------------------------------------------------------------------------
__global__ __launch_bounds__(256) void k_conv1d(
    const float* __restrict__ in, int Cin,
    const float* __restrict__ w, int K, const float* __restrict__ bias,
    const float* __restrict__ g, const float* __restrict__ bb,
    const float* __restrict__ m, const float* __restrict__ v, float eps,
    float* __restrict__ out, int Cout)
{
  int b = blockIdx.x;
  int o = blockIdx.y * 8 + (threadIdx.x >> 5);
  int l = threadIdx.x & 31;
  const float* wr = w + (size_t)o * Cin * K;
  const float* ib = in + (size_t)b * Cin * 32;
  float s = bias[o];
  for (int c = 0; c < Cin; c++)
    for (int j = 0; j < K; j++) {
      int ls = l - (K - 1) + j;
      if (ls >= 0) s += ib[c * 32 + ls] * wr[c * K + j];
    }
  float y = (s - m[o]) * rsqrtf(v[o] + eps) * g[o] + bb[o];
  out[((size_t)b * Cout + o) * 32 + l] = gelu_f(y);
}

// Squeeze-Excite (in place).  grid 32, 256 threads.
__global__ __launch_bounds__(256) void k_se(
    float* __restrict__ data, int C, int Hn,
    const float* __restrict__ w1, const float* __restrict__ w2)
{
  __shared__ float y[256], hh[16];
  int b = blockIdx.x, tid = threadIdx.x;
  float* db = data + (size_t)b * C * 32;
  for (int c = tid; c < C; c += 256) {
    float s = 0.0f;
    for (int l = 0; l < 32; l++) s += db[c * 32 + l];
    y[c] = s * (1.0f / 32.0f);
  }
  __syncthreads();
  if (tid < Hn) {
    float s = 0.0f;
    for (int c = 0; c < C; c++) s += y[c] * w1[tid * C + c];
    hh[tid] = fmaxf(s, 0.0f);
  }
  __syncthreads();
  for (int c = tid; c < C; c += 256) {
    float s = 0.0f;
    for (int k = 0; k < Hn; k++) s += hh[k] * w2[c * Hn + k];
    float sc = sigm(s);
    for (int l = 0; l < 32; l++) db[c * 32 + l] *= sc;
  }
}

__global__ void k_meanL(const float* __restrict__ in, float* __restrict__ out, int C) {
  int b = blockIdx.x, c = threadIdx.x;
  if (c < C) {
    float s = 0.0f;
    for (int l = 0; l < 32; l++) s += in[((size_t)b * C + c) * 32 + l];
    out[b * C + c] = s * (1.0f / 32.0f);
  }
}

// ---------------------------------------------------------------------------
extern "C" void kernel_launch(void* const* d_in, const int* in_sizes, int n_in,
                              void* d_out, int out_size, void* d_ws, size_t ws_size,
                              hipStream_t stream) {
  const float* x    = (const float*)d_in[0];
  const float* we1  = (const float*)d_in[1];
  const float* be1  = (const float*)d_in[2];
  const float* g1   = (const float*)d_in[3];
  const float* b1   = (const float*)d_in[4];
  const float* m1   = (const float*)d_in[5];
  const float* v1   = (const float*)d_in[6];
  const float* we2  = (const float*)d_in[7];
  const float* be2  = (const float*)d_in[8];
  const float* g2   = (const float*)d_in[9];
  const float* b2   = (const float*)d_in[10];
  const float* m2   = (const float*)d_in[11];
  const float* v2   = (const float*)d_in[12];
  const float* w_ih = (const float*)d_in[13];
  const float* w_hh = (const float*)d_in[14];
  const float* b_ih = (const float*)d_in[15];
  const float* b_hh = (const float*)d_in[16];
  const float* muW1 = (const float*)d_in[17];
  const float* muW2 = (const float*)d_in[18];
  const float* muB1 = (const float*)d_in[19];
  const float* muB2 = (const float*)d_in[20];
  const float* stW1 = (const float*)d_in[21];
  const float* stW2 = (const float*)d_in[22];
  const float* stB1 = (const float*)d_in[23];
  const float* stB2 = (const float*)d_in[24];
  const float* mcW1 = (const float*)d_in[25];
  const float* mcW2 = (const float*)d_in[26];
  const float* mcB1 = (const float*)d_in[27];
  const float* mcB2 = (const float*)d_in[28];
  const float* scW1 = (const float*)d_in[29];
  const float* scW2 = (const float*)d_in[30];
  const float* scB1 = (const float*)d_in[31];
  const float* scB2 = (const float*)d_in[32];
  const float* fcW1 = (const float*)d_in[33];
  const float* fcW2 = (const float*)d_in[34];
  const float* fcB1 = (const float*)d_in[35];
  const float* fcB2 = (const float*)d_in[36];
  const float* wc1  = (const float*)d_in[37];
  const float* bc1  = (const float*)d_in[38];
  const float* bn1g = (const float*)d_in[39];
  const float* bn1b = (const float*)d_in[40];
  const float* bn1m = (const float*)d_in[41];
  const float* bn1v = (const float*)d_in[42];
  const float* se1w1 = (const float*)d_in[43];
  const float* se1w2 = (const float*)d_in[44];
  const float* wc2  = (const float*)d_in[45];
  const float* bc2  = (const float*)d_in[46];
  const float* bn2g = (const float*)d_in[47];
  const float* bn2b = (const float*)d_in[48];
  const float* bn2m = (const float*)d_in[49];
  const float* bn2v = (const float*)d_in[50];
  const float* se2w1 = (const float*)d_in[51];
  const float* se2w2 = (const float*)d_in[52];
  const float* wc3  = (const float*)d_in[53];
  const float* bc3  = (const float*)d_in[54];
  const float* bn3g = (const float*)d_in[55];
  const float* bn3b = (const float*)d_in[56];
  const float* bn3m = (const float*)d_in[57];
  const float* bn3v = (const float*)d_in[58];
  const float* eps1 = (const float*)d_in[59];
  const float* eps2 = (const float*)d_in[60];
  (void)in_sizes; (void)n_in; (void)out_size; (void)ws_size;

  // ---- workspace partition ----
  char* w = (char*)d_ws;
  size_t off = 0;
  auto alloc = [&](size_t bytes) { char* p = w + off; off = (off + bytes + 255) & ~(size_t)255; return p; };
  _Float16* we2t  = (_Float16*)alloc(524288 * 2);      // [64][8192]
  _Float16* wihh  = (_Float16*)alloc(32768 * 2);       // [512][64]
  _Float16* whhh  = (_Float16*)alloc(65536 * 2);       // [512][128]
  _Float16* xsh   = (_Float16*)alloc(524288 * 2);      // [8192][64]
  float* gates    = (float*)alloc((size_t)8192 * 512 * 4);
  float* hseq     = (float*)alloc((size_t)32 * 256 * 128 * 4);
  float* sim      = (float*)alloc((size_t)32 * 256 * 256 * 4);
  float* last     = (float*)alloc(4096 * 4);
  float* corr     = (float*)alloc(4096 * 4);
  float* kmu      = (float*)alloc(4096 * 4);
  float* kstd     = (float*)alloc(4096 * 4);
  float* kmuc     = (float*)alloc(4096 * 4);
  float* kstdc    = (float*)alloc(4096 * 4);
  float* x_z      = (float*)alloc(8192 * 4);
  float* x_all    = (float*)alloc(8192 * 4);
  float* c1       = (float*)alloc((size_t)32 * 128 * 32 * 4);
  float* c2       = (float*)alloc((size_t)32 * 256 * 32 * 4);
  float* c3       = (float*)alloc((size_t)32 * 128 * 32 * 4);
  float* x2v      = (float*)alloc(4096 * 4);
  float* outF     = (float*)d_out;                      // [32][10]

  // ---- weight conversions (f32 -> f16, layouts already [N][K]) ----
  k_cvt_h<<<(524288 + 255) / 256, 256, 0, stream>>>(we2, we2t, 524288);
  k_cvt_h<<<(32768 + 255) / 256, 256, 0, stream>>>(w_ih, wihh, 32768);
  k_cvt_h<<<(65536 + 255) / 256, 256, 0, stream>>>(w_hh, whhh, 65536);

  // ---- encoder (fused conv/BN/GELU -> TDM-pipelined WMMA GEMM) ----
  k_encoder<<<dim3(32, 16), 128, 0, stream>>>(x, we1, be1, g1, b1, m1, v1,
                                              we2t, be2, g2, b2, m2, v2, xsh);
  // ---- input-gate GEMM ----
  k_gates_in<<<dim3(512, 8), 128, 0, stream>>>(xsh, wihh, b_ih, b_hh, gates);
  // ---- persistent LSTM scan (w_hh TDM-parked in LDS) ----
  k_lstm<<<1, 256, 0, stream>>>(gates, whhh, hseq);
  // ---- similarity GEMM + clustering ----
  k_sim<<<32, 256, 0, stream>>>(hseq, sim);
  k_cluster<<<32, 256, 0, stream>>>(sim, hseq, last, corr);
  // ---- KAN heads on corr ----
  k_kan<<<dim3(32, 128), 128, 0, stream>>>(corr, 128, 128, muW1, muW2, muB1, muB2, kmu);
  k_kan<<<dim3(32, 128), 128, 0, stream>>>(corr, 128, 128, stW1, stW2, stB1, stB2, kstd);
  k_zcat<<<16, 256, 0, stream>>>(kmu, kstd, eps1, last, x_z);
  k_kan<<<dim3(32, 128), 128, 0, stream>>>(x_z, 256, 128, mcW1, mcW2, mcB1, mcB2, kmuc);
  k_kan<<<dim3(32, 128), 128, 0, stream>>>(x_z, 256, 128, scW1, scW2, scB1, scB2, kstdc);
  // ---- causal conv / SE branch (input x reinterpreted as [B][256][32]) ----
  k_conv1d<<<dim3(32, 16), 256, 0, stream>>>(x, 256, wc1, 8, bc1,
                                             bn1g, bn1b, bn1m, bn1v, 1e-3f, c1, 128);
  k_se<<<32, 256, 0, stream>>>(c1, 128, 8, se1w1, se1w2);
  k_conv1d<<<dim3(32, 32), 256, 0, stream>>>(c1, 128, wc2, 5, bc2,
                                             bn2g, bn2b, bn2m, bn2v, 1e-3f, c2, 256);
  k_se<<<32, 256, 0, stream>>>(c2, 256, 16, se2w1, se2w2);
  k_conv1d<<<dim3(32, 16), 256, 0, stream>>>(c2, 256, wc3, 3, bc3,
                                             bn3g, bn3b, bn3m, bn3v, 1e-3f, c3, 128);
  k_meanL<<<32, 128, 0, stream>>>(c3, x2v, 128);
  // ---- combine + final KAN head -> d_out ----
  k_x1cat<<<16, 256, 0, stream>>>(kmuc, kstdc, eps2, x2v, x_all);
  k_kan<<<dim3(32, 10), 128, 0, stream>>>(x_all, 256, 10, fcW1, fcW2, fcB1, fcB2, outF);
}
